// CapsNet_18098992185787
// MI455X (gfx1250) — matrix operand order repaired and checked by reading, not compiled, not run on hardware
//
#include <hip/hip_runtime.h>
#include <hip/hip_bf16.h>
#include <stdint.h>

// ---------------- types ----------------
typedef __attribute__((ext_vector_type(16))) __bf16        bf16x16;
typedef __attribute__((ext_vector_type(8)))  float         f32x8;
typedef __attribute__((ext_vector_type(4)))  unsigned int  u32x4;

union FragCast { u32x4 q[2]; bf16x16 v; };

__device__ __forceinline__ unsigned short f32_to_bf16(float f) {
    unsigned int x = __builtin_bit_cast(unsigned int, f);
    unsigned int r = x + 0x7FFFu + ((x >> 16) & 1u);   // round-to-nearest-even
    return (unsigned short)(r >> 16);
}
__device__ __forceinline__ float bf16_to_f32(unsigned short h) {
    unsigned int x = ((unsigned int)h) << 16;
    return __builtin_bit_cast(float, x);
}

// gfx1250 async global->LDS copy (ASYNCcnt-tracked, no VGPR staging)
__device__ __forceinline__ void async_load_b128(unsigned lds_addr, const void* gaddr) {
    asm volatile("global_load_async_to_lds_b128 %0, %1, off"
                 :: "v"(lds_addr), "v"(gaddr)
                 : "memory");
}

#if __has_builtin(__builtin_amdgcn_s_wait_asynccnt)
#define WAIT_ASYNC_6() __builtin_amdgcn_s_wait_asynccnt(6)
#else
#define WAIT_ASYNC_6() asm volatile("s_wait_asynccnt 6" ::: "memory")
#endif

// ---------------- problem constants ----------------
#define BATCH   512
#define KTOT    20736         // 81 * 256
#define MTOT    18432         // 512 * 36
#define C2      256
#define NIC     1152
#define NOC     10

// ============================================================
// Kernel 1: conv1 (9x9, 1->256, stride 1) + ReLU -> NHWC bf16
// ============================================================
__global__ __launch_bounds__(256) void conv1_relu_nhwc_bf16(
    const float* __restrict__ x,      // [512,1,28,28]
    const float* __restrict__ w,      // [256,1,9,9]
    const float* __restrict__ bvec,   // [256]
    unsigned short* __restrict__ h1)  // [512,20,20,256] bf16
{
    __shared__ __align__(16) float patch[9 * 28];
    const int t   = threadIdx.x;
    const int blk = blockIdx.x;
    const int b   = blk / 20;
    const int oh  = blk - b * 20;

    if (t < 252) {
        const int row = t / 28, col = t - row * 28;
        patch[t] = x[(size_t)b * 784 + (size_t)(oh + row) * 28 + col];
    }
    float wr[81];
    #pragma unroll
    for (int k = 0; k < 81; ++k) wr[k] = w[(size_t)t * 81 + k];
    const float bias = bvec[t];
    __syncthreads();

    #pragma unroll 1
    for (int ow = 0; ow < 20; ++ow) {
        float acc = bias;
        #pragma unroll
        for (int kh = 0; kh < 9; ++kh)
            #pragma unroll
            for (int kw = 0; kw < 9; ++kw)
                acc = fmaf(patch[kh * 28 + ow + kw], wr[kh * 9 + kw], acc);
        acc = fmaxf(acc, 0.0f);
        h1[(((size_t)b * 20 + oh) * 20 + ow) * 256 + t] = f32_to_bf16(acc);
    }
}

// ============================================================
// Kernel 2: reformat conv2 weights -> W2T [N=256][K=20736] bf16
//           with k = (kh*9+kw)*256 + c
// ============================================================
__global__ __launch_bounds__(256) void w2_reformat(
    const float* __restrict__ w2,         // [256,256,9,9]
    unsigned short* __restrict__ w2t)     // [256,20736] bf16
{
    const int t = blockIdx.x * 256 + threadIdx.x;
    if (t >= 256 * KTOT) return;
    const int n    = t / KTOT;
    const int rest = t - n * KTOT;
    const int pos  = rest >> 8;     // kh*9+kw
    const int c    = rest & 255;
    w2t[t] = f32_to_bf16(w2[(size_t)(n * 256 + c) * 81 + pos]);
}

// ============================================================
// Kernel 3: conv2 as implicit GEMM via bf16 WMMA (f32 accum).
//   M=18432  N=256  K=20736.  Block tile 128x64, K-stage 64.
//   Triple-buffered LDS fed by GLOBAL_LOAD_ASYNC_TO_LDS_B128
//   (ASYNCcnt), ONE barrier per stage. 8 waves: 4(M) x 2(N),
//   each wave computes 2x2 16x16 tiles (max fragment reuse).
// ============================================================
#define BM 128
#define BN 64
#define KS 64
#define ABYTES (BM * KS * 2)   // 16 KB per A buffer
#define BBYTES (BN * KS * 2)   //  8 KB per B buffer

__global__ __launch_bounds__(256) void conv2_gemm_wmma(
    const unsigned short* __restrict__ h1,   // [512,20,20,256] bf16 NHWC
    const unsigned short* __restrict__ w2t,  // [256][20736] bf16
    const float* __restrict__ c2b,           // [256]
    float* __restrict__ u)                   // [512][1152][8] f32
{
    __shared__ __align__(16) unsigned short As[3][BM * KS];  // 3 x 16 KB
    __shared__ __align__(16) unsigned short Bs[3][BN * KS];  // 3 x  8 KB

    const int t    = threadIdx.x;
    const int lane = t & 31;
    const int wv   = t >> 5;
    const int mg   = wv >> 1;        // 0..3 : 32-row group
    const int ng   = wv & 1;         // 0..1 : 32-col group
    const int bx   = blockIdx.x;
    const int by   = blockIdx.y;

    // ---- A loader mapping: thread -> (row, half of 64-elem K slice)
    const int ar   = t >> 1;
    const int ah   = t & 1;
    const int am   = bx * BM + ar;
    const int ab   = am / 36;
    const int arem = am - ab * 36;
    const int aoh  = arem / 6;
    const int aow  = arem - aoh * 6;
    const size_t abase = ((size_t)(ab * 20 + aoh * 2) * 20 + aow * 2) * 256 + ah * 32;

    // ---- B loader mapping: thread -> (n row, quarter of K slice)
    const int bn = t >> 2;
    const int bq = t & 3;
    const size_t bbase = (size_t)(by * BN + bn) * KTOT + bq * 16;

    // LDS byte addresses for this thread's async destinations
    const unsigned aLds0 = (unsigned)(size_t)&As[0][0] + (unsigned)(ar * KS + ah * 32) * 2;
    const unsigned bLds0 = (unsigned)(size_t)&Bs[0][0] + (unsigned)(bn * KS + bq * 16) * 2;

    const f32x8 zero = {0.f, 0.f, 0.f, 0.f, 0.f, 0.f, 0.f, 0.f};
    f32x8 acc00 = zero, acc01 = zero, acc10 = zero, acc11 = zero;

    const int NSTAGE = KTOT / KS;   // 324

    // ---- prologue: stage 0 -> buffer 0
    {
        const unsigned short* ap = h1 + abase;     // kh=kw=0, c0=0
        #pragma unroll
        for (int c = 0; c < 4; ++c) async_load_b128(aLds0 + c * 16, ap + c * 8);
        const unsigned short* bp = w2t + bbase;
        async_load_b128(bLds0,      bp);
        async_load_b128(bLds0 + 16, bp + 8);
    }

    for (int s = 0; s < NSTAGE; ++s) {
        // ---- issue async loads for stage s+1 into buffer (s+1)%3
        // (last iteration re-issues the final stage into a dead buffer so the
        //  asynccnt wait threshold stays a compile-time constant)
        {
            const int sn  = (s + 1 < NSTAGE) ? (s + 1) : (NSTAGE - 1);
            const int nb  = (s + 1) % 3;
            const int k0  = sn * KS;
            const int pos = k0 >> 8;
            const int c0  = k0 & 255;
            const int kh  = pos / 9;
            const int kw  = pos - kh * 9;
            const unsigned aL = aLds0 + (unsigned)(nb * ABYTES);
            const unsigned bL = bLds0 + (unsigned)(nb * BBYTES);
            const unsigned short* ap = h1 + abase + (size_t)(kh * 20 + kw) * 256 + c0;
            #pragma unroll
            for (int c = 0; c < 4; ++c) async_load_b128(aL + c * 16, ap + c * 8);
            const unsigned short* bp = w2t + bbase + k0;
            async_load_b128(bL,      bp);
            async_load_b128(bL + 16, bp + 8);
        }
        // stage s data resident once ASYNCcnt <= 6 (in-order completion)
        WAIT_ASYNC_6();
        __syncthreads();

        // ---- compute from buffer s%3: 2 K-halves x (2x2 tiles) = 8 WMMAs
        const int cb = s % 3;
        const unsigned short* Ab = &As[cb][0];
        const unsigned short* Bb = &Bs[cb][0];
        #pragma unroll
        for (int kb = 0; kb < 2; ++kb) {
            FragCast a0, a1, b0, b1;
            // A 16x32 bf16 layout: lanes<16 K{0..7,16..23}; lanes>=16 K{8..15,24..31}
            const unsigned short* ap0 = Ab + (mg * 32 + (lane & 15)) * KS + kb * 32 + (lane >> 4) * 8;
            const unsigned short* ap1 = ap0 + 16 * KS;
            a0.q[0] = *(const u32x4*)(ap0);  a0.q[1] = *(const u32x4*)(ap0 + 16);
            a1.q[0] = *(const u32x4*)(ap1);  a1.q[1] = *(const u32x4*)(ap1 + 16);
            // B 32x16: lane = N col; lanes<16 K{0..7,16..23}; lanes>=16 K{8..15,24..31}
            const unsigned short* bp0 = Bb + (ng * 32 + (lane & 15)) * KS + kb * 32 + (lane >> 4) * 8;
            const unsigned short* bp1 = bp0 + 16 * KS;
            b0.q[0] = *(const u32x4*)(bp0);  b0.q[1] = *(const u32x4*)(bp0 + 16);
            b1.q[0] = *(const u32x4*)(bp1);  b1.q[1] = *(const u32x4*)(bp1 + 16);
            acc00 = __builtin_amdgcn_wmma_f32_16x16x32_bf16(false, a0.v, false, b0.v, (short)0, acc00, false, false);
            acc01 = __builtin_amdgcn_wmma_f32_16x16x32_bf16(false, a0.v, false, b1.v, (short)0, acc01, false, false);
            acc10 = __builtin_amdgcn_wmma_f32_16x16x32_bf16(false, a1.v, false, b0.v, (short)0, acc10, false, false);
            acc11 = __builtin_amdgcn_wmma_f32_16x16x32_bf16(false, a1.v, false, b1.v, (short)0, acc11, false, false);
        }
    }

    // ---- epilogue: bias + scatter into capsule layout u[b][i][p]
    // C/D layout: VGPR r -> M = r (+8 for lanes>=16); N = lane&15
    f32x8 accs[2][2] = {{acc00, acc01}, {acc10, acc11}};
    #pragma unroll
    for (int mt = 0; mt < 2; ++mt) {
        const int mbase = bx * BM + mg * 32 + mt * 16 + ((lane >> 4) * 8);
        #pragma unroll
        for (int nt = 0; nt < 2; ++nt) {
            const int n    = by * BN + ng * 32 + nt * 16 + (lane & 15);
            const float bs = c2b[n];
            #pragma unroll
            for (int r = 0; r < 8; ++r) {
                const int m   = mbase + r;
                const int b   = m / 36;
                const int rem = m - b * 36;
                const int oh  = rem / 6;
                const int ow  = rem - oh * 6;
                const int i   = ow * 192 + oh * 32 + (n >> 3);   // transpose(0,3,2,1)
                u[((size_t)b * NIC + i) * 8 + (n & 7)] = accs[mt][nt][r] + bs;
            }
        }
    }
}

// ============================================================
// Kernel 4: u_hat[b,i,j,q] = sum_p u[b,i,p] * capW[i,j,p,q] -> bf16
// ============================================================
__global__ __launch_bounds__(256) void uhat_kernel(
    const float* __restrict__ u,          // [512,1152,8]
    const float* __restrict__ capW,       // [1152,10,8,16]
    unsigned short* __restrict__ uhat)    // [512,1152,10,16] bf16
{
    const int i = blockIdx.x;
    const int t = threadIdx.x;
    __shared__ __align__(16) float Wl[1280];
    __shared__ __align__(16) float ul[32 * 8];

    for (int x = t; x < 1280; x += 256) Wl[x] = capW[(size_t)i * 1280 + x];
    const int j = t >> 4, q = t & 15;

    for (int chunk = 0; chunk < 16; ++chunk) {
        const int b0 = chunk * 32;
        __syncthreads();
        ul[t] = u[((size_t)(b0 + (t >> 3)) * NIC + i) * 8 + (t & 7)];
        __syncthreads();
        if (t < 160) {
            #pragma unroll 4
            for (int bb = 0; bb < 32; ++bb) {
                float acc = 0.f;
                #pragma unroll
                for (int p = 0; p < 8; ++p)
                    acc = fmaf(ul[bb * 8 + p], Wl[j * 128 + p * 16 + q], acc);
                uhat[((size_t)(b0 + bb) * NIC + i) * 160 + j * 16 + q] = f32_to_bf16(acc);
            }
        }
    }
}

// ============================================================
// Kernel 5: dynamic routing (3 iters) + squash -> v [512,10,16] f32
// ============================================================
__global__ __launch_bounds__(256) void routing_kernel(
    const unsigned short* __restrict__ uhat,  // [512,1152,10,16] bf16
    float* __restrict__ out)                  // [512,10,16]
{
    const int b = blockIdx.x;
    const int t = threadIdx.x;
    __shared__ float cl[NIC * NOC];   // coupling coefficients (46 KB)
    __shared__ float s_sh[160];
    __shared__ float f_sh[10];
    __shared__ float v_sh[160];

    const int nrows = (t < 128) ? 5 : 4;      // 1152 = 4*256 + 128
    float breg[5][10];
    #pragma unroll
    for (int r = 0; r < 5; ++r)
        #pragma unroll
        for (int jj = 0; jj < 10; ++jj) breg[r][jj] = 0.f;

    const size_t ubase = (size_t)b * (NIC * 160);
    const int j = t >> 4, q = t & 15;

    for (int it = 0; it < 3; ++it) {
        // ---- c = softmax_j(b_ij) -> LDS
        for (int r = 0; r < nrows; ++r) {
            const int i = t + 256 * r;
            float m = breg[r][0];
            #pragma unroll
            for (int jj = 1; jj < 10; ++jj) m = fmaxf(m, breg[r][jj]);
            float e[10], sum = 0.f;
            #pragma unroll
            for (int jj = 0; jj < 10; ++jj) { e[jj] = __expf(breg[r][jj] - m); sum += e[jj]; }
            const float inv = 1.0f / sum;
            #pragma unroll
            for (int jj = 0; jj < 10; ++jj) cl[i * 10 + jj] = e[jj] * inv;
        }
        __syncthreads();
        // ---- s_jq = sum_i c_ij * uhat_ijq (u_hat streamed from L2)
        if (t < 160) {
            float s = 0.f;
            const unsigned short* up = uhat + ubase + j * 16 + q;
            for (int i = 0; i < NIC; ++i)
                s = fmaf(cl[i * 10 + j], bf16_to_f32(up[(size_t)i * 160]), s);
            s_sh[t] = s;
        }
        __syncthreads();
        // ---- squash factor per output capsule
        if (t < 10) {
            float sq = 0.f;
            #pragma unroll
            for (int qq = 0; qq < 16; ++qq) { float v = s_sh[t * 16 + qq]; sq = fmaf(v, v, sq); }
            f_sh[t] = (sq / (1.0f + sq)) / sqrtf(sq + 1e-8f);
        }
        __syncthreads();
        if (t < 160) v_sh[t] = s_sh[t] * f_sh[j];
        __syncthreads();
        // ---- b_ij += <uhat_ij, v_j>  (skipped after final v)
        if (it < 2) {
            for (int r = 0; r < nrows; ++r) {
                const int i = t + 256 * r;
                const unsigned short* up = uhat + ubase + (size_t)i * 160;
                #pragma unroll
                for (int jj = 0; jj < 10; ++jj) {
                    float dot = 0.f;
                    #pragma unroll
                    for (int qq = 0; qq < 16; ++qq)
                        dot = fmaf(bf16_to_f32(up[jj * 16 + qq]), v_sh[jj * 16 + qq], dot);
                    breg[r][jj] += dot;
                }
            }
        }
    }
    if (t < 160) out[(size_t)b * 160 + t] = v_sh[t];
}

// ============================================================
// host launcher
// ============================================================
extern "C" void kernel_launch(void* const* d_in, const int* in_sizes, int n_in,
                              void* d_out, int out_size, void* d_ws, size_t ws_size,
                              hipStream_t stream) {
    const float* x    = (const float*)d_in[0];   // [512,1,28,28]
    const float* w1   = (const float*)d_in[1];   // [256,1,9,9]
    const float* b1   = (const float*)d_in[2];   // [256]
    const float* w2   = (const float*)d_in[3];   // [256,256,9,9]
    const float* b2   = (const float*)d_in[4];   // [256]
    const float* capW = (const float*)d_in[5];   // [1152,10,8,16]

    char* ws = (char*)d_ws;
    // workspace layout (bytes):
    //   h1  bf16 [512,20,20,256]      : 104,857,600
    //   w2t bf16 [256,20736]          :  10,616,832
    //   u   f32  [512,1152,8]         :  18,874,368
    //   uhat bf16 [512,1152,10,16]    : 188,743,680  (fits MI455X 192MB L2)
    unsigned short* h1  = (unsigned short*)(ws);
    unsigned short* w2t = (unsigned short*)(ws + 104857600ull);
    float*          u   = (float*)(ws + 115474432ull);
    unsigned short* uh  = (unsigned short*)(ws + 134348800ull);

    hipLaunchKernelGGL(conv1_relu_nhwc_bf16, dim3(BATCH * 20), dim3(256), 0, stream,
                       x, w1, b1, h1);
    hipLaunchKernelGGL(w2_reformat, dim3((256 * KTOT + 255) / 256), dim3(256), 0, stream,
                       w2, w2t);
    hipLaunchKernelGGL(conv2_gemm_wmma, dim3(MTOT / BM, C2 / BN), dim3(256), 0, stream,
                       h1, w2t, b2, u);
    hipLaunchKernelGGL(uhat_kernel, dim3(NIC), dim3(256), 0, stream,
                       u, capW, uh);
    hipLaunchKernelGGL(routing_kernel, dim3(BATCH), dim3(256), 0, stream,
                       uh, (float*)d_out);
}
